// SelfAttention_49460843381298
// MI455X (gfx1250) — matrix operand order repaired and checked by reading, not compiled, not run on hardware
//
#include <hip/hip_runtime.h>

// ---------------------------------------------------------------------------
// MI455X (gfx1250) self-attention, bf16 WMMA pipeline, f32 accumulate/softmax.
// Reference: per-head Q/K/V projections, energies = QK^T + AP, mask -> -1e9,
// softmax over k, y = A^T V (transposed attention), out = y Wout^T.
// Outputs: out [4,2048,1024] f32, then A[:,H-1] [4,2048,2048] f32.
// ---------------------------------------------------------------------------

typedef __bf16 bf16;
typedef __attribute__((ext_vector_type(16))) __bf16 v16bf;
typedef __attribute__((ext_vector_type(8)))  __bf16 v8bf;
typedef __attribute__((ext_vector_type(8)))  float  v8f;

constexpr int BS  = 4;
constexpr int N   = 2048;
constexpr int D   = 1024;
constexpr int H   = 8;
constexpr int DH  = 128;    // D / H
constexpr int OUT = 1024;
constexpr float NEG_INF = -1e9f;
constexpr float LOG_FREQ = 9.210340371976184f; // ln(10000)

#if defined(__AMDGCN__) && __has_builtin(__builtin_amdgcn_global_load_async_to_lds_b128)
#define HAVE_ASYNC_LDS 1
// Builtin signature (from hipcc diagnostic): param 1 is v4i in addrspace(1),
// param 2 is the LDS destination (addrspace(3)).
typedef __attribute__((vector_size(16))) int v4i_async;
typedef __attribute__((address_space(1))) v4i_async* gas1_v4i;
typedef __attribute__((address_space(3))) v4i_async* las3_v4i;
#else
#define HAVE_ASYNC_LDS 0
#endif

// D = A(16x32 bf16) * B(32x16 bf16) + C(16x16 f32), wave32.
__device__ inline v8f wmma_bf16(v16bf a, v16bf b, v8f c) {
  return __builtin_amdgcn_wmma_f32_16x16x32_bf16(
      /*neg_a=*/false, a, /*neg_b=*/false, b,
      /*c_mod=*/(short)0, c, /*reuse_a=*/false, /*reuse_b=*/false);
}

__device__ inline v8f zero8() {
  v8f z;
#pragma unroll
  for (int i = 0; i < 8; ++i) z[i] = 0.0f;
  return z;
}

// A fragment, 16x32 (MxK), row-major source with leading dim ld.
// ISA layout: lanes 0-15 row M=lane, K={0..7,16..23}; lanes 16-31 row M=lane-16,
// K={8..15,24..31}.  Two 16B loads per lane.
__device__ inline v16bf load_a_frag(const bf16* base, int ld, int row0, int k0, int lane) {
  const int m = lane & 15, hh = lane >> 4;
  const bf16* p = base + (size_t)(row0 + m) * ld + k0 + hh * 8;
  v8bf lo = *(const v8bf*)p;
  v8bf hi = *(const v8bf*)(p + 16);
  v16bf a;
#pragma unroll
  for (int t = 0; t < 8; ++t) { a[t] = lo[t]; a[8 + t] = hi[t]; }
  return a;
}

// B fragment, 32x16 (KxN) where B[k,n] = W[n0+n, k0+k], W row-major lead dim ld.
// ISA layout: lanes 0-15 col N=lane hold K={0..7,16..23}; lanes 16-31 col N=lane-16
// hold K={8..15,24..31}. Two 16B loads per lane.
__device__ inline v16bf load_b_frag(const bf16* base, int ld, int n0, int k0, int lane) {
  const int n = lane & 15, hh = lane >> 4;
  const bf16* p = base + (size_t)(n0 + n) * ld + k0 + hh * 8;
  v8bf lo = *(const v8bf*)p;
  v8bf hi = *(const v8bf*)(p + 16);
  v16bf b;
#pragma unroll
  for (int t = 0; t < 8; ++t) { b[t] = lo[t]; b[8 + t] = hi[t]; }
  return b;
}

// ---------------------------------------------------------------------------
// K0: f32 -> bf16 conversions for x and all weight matrices.
// ---------------------------------------------------------------------------
__global__ void __launch_bounds__(256)
convert_kernel(const float* __restrict__ x,  const float* __restrict__ wq,
               const float* __restrict__ wk, const float* __restrict__ wv,
               const float* __restrict__ wo,
               bf16* xb, bf16* wqb, bf16* wkb, bf16* wvb, bf16* wob) {
  const size_t tid = (size_t)blockIdx.x * blockDim.x + threadIdx.x;
  const size_t stride = (size_t)gridDim.x * blockDim.x;
  const size_t nx = (size_t)BS * N * D;
  for (size_t i = tid; i < nx; i += stride) xb[i] = (bf16)x[i];
  const size_t nw = (size_t)H * DH * D;
  for (size_t i = tid; i < nw; i += stride) {
    wqb[i] = (bf16)wq[i]; wkb[i] = (bf16)wk[i]; wvb[i] = (bf16)wv[i];
  }
  const size_t no = (size_t)OUT * D;
  for (size_t i = tid; i < no; i += stride) wob[i] = (bf16)wo[i];
}

// ---------------------------------------------------------------------------
// K1: AP[p,2i]=sin(p/freq^(2i/D)), AP[p,2i+1]=cos(...), shared across (b,h).
// ---------------------------------------------------------------------------
__global__ void __launch_bounds__(256)
ap_kernel(float* __restrict__ ap) {
  const size_t tid = (size_t)blockIdx.x * blockDim.x + threadIdx.x;
  const size_t stride = (size_t)gridDim.x * blockDim.x;
  const size_t total = (size_t)N * N;
  for (size_t i = tid; i < total; i += stride) {
    const int j = (int)(i >> 11);       // N = 2048
    const int k = (int)(i & (N - 1));
    const int i2 = k >> 1;
    const float inv = __expf(-LOG_FREQ * (2.0f * (float)i2) * (1.0f / (float)D));
    const float ang = (float)j * inv;
    ap[i] = (k & 1) ? __cosf(ang) : __sinf(ang);
  }
}

// ---------------------------------------------------------------------------
// K2: Q/K/V projections, register-tiled: each wave computes a 32x64 output
// block (2 row-frags x 4 col-frags -> 8 WMMAs per 32-deep k-step) to raise
// L1 arithmetic intensity ~2.7x over a single 16x16 tile per wave.
// ---------------------------------------------------------------------------
__global__ void __launch_bounds__(256)
qkv_proj_kernel(const bf16* __restrict__ xb,
                const bf16* __restrict__ wqb, const bf16* __restrict__ wkb,
                const bf16* __restrict__ wvb,
                bf16* __restrict__ Qb, bf16* __restrict__ Kb, bf16* __restrict__ Vb) {
  const int lane = threadIdx.x & 31;
  int gw = blockIdx.x * 8 + (threadIdx.x >> 5);
  const int cg = gw & 1;  gw >>= 1;   // 64-wide col group (DH/64 = 2)
  const int rt = gw & 63; gw >>= 6;   // 32-tall row tile  (N/32 = 64)
  const int h  = gw & 7;  gw >>= 3;
  const int b  = gw & 3;  gw >>= 2;
  const int which = gw;               // 0=Q 1=K 2=V

  const bf16* wsel = (which == 0) ? wqb : (which == 1) ? wkb : wvb;
  bf16*       osel = (which == 0) ? Qb  : (which == 1) ? Kb  : Vb;
  const bf16* abase = xb + (size_t)b * N * D;
  const bf16* wbase = wsel + (size_t)h * DH * D;
  bf16* obase = osel + (size_t)(b * H + h) * N * DH;

  const int row0 = rt * 32, n0 = cg * 64;
  v8f c[2][4];
#pragma unroll
  for (int mi = 0; mi < 2; ++mi)
#pragma unroll
    for (int ni = 0; ni < 4; ++ni) c[mi][ni] = zero8();

  for (int kc = 0; kc < D / 32; ++kc) {
    if (kc + 1 < D / 32)
      __builtin_prefetch(abase + (size_t)row0 * D + (kc + 1) * 32, 0, 0);
    v16bf a0 = load_a_frag(abase, D, row0,      kc * 32, lane);
    v16bf a1 = load_a_frag(abase, D, row0 + 16, kc * 32, lane);
#pragma unroll
    for (int ni = 0; ni < 4; ++ni) {
      v16bf bb = load_b_frag(wbase, D, n0 + ni * 16, kc * 32, lane);
      c[0][ni] = wmma_bf16(a0, bb, c[0][ni]);
      c[1][ni] = wmma_bf16(a1, bb, c[1][ni]);
    }
  }
  const int m = lane & 15, hh = lane >> 4;
#pragma unroll
  for (int mi = 0; mi < 2; ++mi)
#pragma unroll
    for (int ni = 0; ni < 4; ++ni)
#pragma unroll
      for (int r = 0; r < 8; ++r)
        obase[(size_t)(row0 + mi * 16 + hh * 8 + r) * DH + n0 + ni * 16 + m] =
            (bf16)c[mi][ni][r];
}

// ---------------------------------------------------------------------------
// K3: softmax row statistics (flash-style): per (b,h,j) rowmax M and
// rowsum L of exp(e - M), with e = mask ? QK^T + AP : -1e9.
// One wave owns a 16-row strip (Q frags held in registers for all 128
// k-tiles); row-wise reductions are 16-lane butterflies.
// ---------------------------------------------------------------------------
__global__ void __launch_bounds__(256)
pass1_stats_kernel(const bf16* __restrict__ Qb, const bf16* __restrict__ Kb,
                   const float* __restrict__ ap, const int* __restrict__ mask,
                   float* __restrict__ Mrow, float* __restrict__ Lrow) {
  const int lane = threadIdx.x & 31, w = threadIdx.x >> 5;
  const int bh = blockIdx.x >> 4;
  const int b  = bh >> 3;
  const int j0 = (blockIdx.x & 15) * 128 + w * 16;
  const bf16* qb = Qb + (size_t)bh * N * DH;
  const bf16* kb = Kb + (size_t)bh * N * DH;
  const int m = lane & 15, hh = lane >> 4;

  v16bf qf[4];
#pragma unroll
  for (int ec = 0; ec < 4; ++ec) qf[ec] = load_a_frag(qb, DH, j0, ec * 32, lane);

  int mj[8]; float mx[8], sm[8];
#pragma unroll
  for (int r = 0; r < 8; ++r) {
    mj[r] = mask[b * N + j0 + hh * 8 + r];
    mx[r] = -3.0e38f; sm[r] = 0.0f;
  }

  for (int kt = 0; kt < N / 16; ++kt) {
    const int k0 = kt * 16;
    v8f s = zero8();
#pragma unroll
    for (int ec = 0; ec < 4; ++ec) {
      v16bf bb = load_b_frag(kb, DH, k0, ec * 32, lane);
      s = wmma_bf16(qf[ec], bb, s);
    }
    const int mk = mask[b * N + k0 + m];
#pragma unroll
    for (int r = 0; r < 8; ++r) {
      const int j = j0 + hh * 8 + r;
      const float e = (mj[r] && mk) ? (s[r] + ap[(size_t)j * N + k0 + m]) : NEG_INF;
      float rm = e;
      rm = fmaxf(rm, __shfl_xor(rm, 1, 16));
      rm = fmaxf(rm, __shfl_xor(rm, 2, 16));
      rm = fmaxf(rm, __shfl_xor(rm, 4, 16));
      rm = fmaxf(rm, __shfl_xor(rm, 8, 16));
      const float mn = fmaxf(mx[r], rm);
      float p = __expf(e - mn);
      p += __shfl_xor(p, 1, 16);
      p += __shfl_xor(p, 2, 16);
      p += __shfl_xor(p, 4, 16);
      p += __shfl_xor(p, 8, 16);
      sm[r] = sm[r] * __expf(mx[r] - mn) + p;
      mx[r] = mn;
    }
  }
  if (m == 0) {
#pragma unroll
    for (int r = 0; r < 8; ++r) {
      const int j = j0 + hh * 8 + r;
      Mrow[(size_t)bh * N + j] = mx[r];
      Lrow[(size_t)bh * N + j] = sm[r];
    }
  }
}

// ---------------------------------------------------------------------------
// K4: recompute scores tile-wise, normalize -> P, emit last-head A,
// accumulate y = P^T V via WMMA.  Workgroup = (b,h, 128-wide i block);
// wave w owns i columns [i0, i0+16) and all dh=128 e columns (8 C tiles).
// V tile staged to LDS via gfx1250 async-DMA when available (ASYNCcnt path),
// then transposed LDS->LDS; P transposed through per-wave LDS tile (bf16).
// ---------------------------------------------------------------------------
__global__ void __launch_bounds__(256)
pass2_attn_kernel(const bf16* __restrict__ Qb, const bf16* __restrict__ Kb,
                  const bf16* __restrict__ Vb, const float* __restrict__ ap,
                  const int* __restrict__ mask,
                  const float* __restrict__ Mrow, const float* __restrict__ Lrow,
                  bf16* __restrict__ yb, float* __restrict__ aout) {
  __shared__ alignas(16) bf16 VT[DH][32];     // V^T tile: [e][j in 0..31]
  __shared__ alignas(16) bf16 PT[8][16][32];  // per-wave P^T tile: [i][j in 0..31]
#if HAVE_ASYNC_LDS
  __shared__ alignas(16) bf16 VR[32][DH];     // async-DMA landing: V rows
#endif

  const int tid = threadIdx.x;
  const int lane = tid & 31, w = tid >> 5;
  const int bh = blockIdx.x >> 4;
  const int b  = bh >> 3, h = bh & 7;
  const int i0 = (blockIdx.x & 15) * 128 + w * 16;
  const int m = lane & 15, hh = lane >> 4;

  const bf16* qb = Qb + (size_t)bh * N * DH;
  const bf16* kb = Kb + (size_t)bh * N * DH;
  const bf16* vb = Vb + (size_t)bh * N * DH;
  const float* Mb = Mrow + (size_t)bh * N;
  const float* Lb = Lrow + (size_t)bh * N;

  // B fragments of K rows i0..i0+15 (columns of K^T), reused for every j tile.
  v16bf kf[4];
#pragma unroll
  for (int ec = 0; ec < 4; ++ec) kf[ec] = load_b_frag(kb, DH, i0, ec * 32, lane);
  const int mki = mask[b * N + i0 + m];

  v8f acc[8];
#pragma unroll
  for (int et = 0; et < 8; ++et) acc[et] = zero8();

  for (int j0 = 0; j0 < N; j0 += 32) {
    __syncthreads();
#if HAVE_ASYNC_LDS
    // Async DMA V[j0:j0+32, 0:128] row-major into LDS (512 x 16B chunks),
    // then transpose LDS->LDS into VT.
#pragma unroll
    for (int t = 0; t < 2; ++t) {
      const int idx = t * 256 + tid;
      const int jj = idx >> 4;            // 16 chunks per 256B row
      const int ee = (idx & 15) * 8;
      __builtin_amdgcn_global_load_async_to_lds_b128(
          (gas1_v4i)(vb + (size_t)(j0 + jj) * DH + ee),
          (las3_v4i)(&VR[jj][ee]), 0, 0);
    }
#if __has_builtin(__builtin_amdgcn_s_wait_asynccnt)
    __builtin_amdgcn_s_wait_asynccnt(0);
#else
    asm volatile("s_wait_asynccnt 0x0" ::: "memory");
#endif
    __syncthreads();
#pragma unroll
    for (int t = 0; t < 16; ++t) {
      const int idx = t * 256 + tid;
      const int jj = idx >> 7, e = idx & 127;
      VT[e][jj] = VR[jj][e];
    }
#else
    // Stage V[j0:j0+32, 0:128] transposed into LDS (coalesced global reads).
#pragma unroll
    for (int t = 0; t < 16; ++t) {
      const int idx = t * 256 + tid;
      const int jj = idx >> 7, e = idx & 127;
      VT[e][jj] = vb[(size_t)(j0 + jj) * DH + e];
    }
#endif
    __syncthreads();

#pragma unroll
    for (int hs = 0; hs < 2; ++hs) {
      const int js = j0 + hs * 16;
      v8f s = zero8();
#pragma unroll
      for (int ec = 0; ec < 4; ++ec) {
        v16bf a = load_a_frag(qb, DH, js, ec * 32, lane);
        s = wmma_bf16(a, kf[ec], s);
      }
#pragma unroll
      for (int r = 0; r < 8; ++r) {
        const int j = js + hh * 8 + r;
        const float Mv = Mb[j], Lv = Lb[j];
        const float e_ = (mask[b * N + j] && mki)
                             ? (s[r] + ap[(size_t)j * N + i0 + m]) : NEG_INF;
        const float p = __expf(e_ - Mv) / Lv;
        if (h == H - 1)
          aout[((size_t)b * N + j) * N + i0 + m] = p;
        PT[w][m][hs * 16 + hh * 8 + r] = (bf16)p;   // transpose to [i][j]
      }
    }

    // A fragment of P^T (M=i, K=j in 0..31) from per-wave LDS tile.
    v16bf pa;
    {
      const bf16* pp = &PT[w][m][hh * 8];
      v8bf lo = *(const v8bf*)pp;
      v8bf hi = *(const v8bf*)(pp + 16);
#pragma unroll
      for (int t = 0; t < 8; ++t) { pa[t] = lo[t]; pa[8 + t] = hi[t]; }
    }
    // y += P^T * V over all 8 e tiles; B frag from staged V^T.
#pragma unroll
    for (int et = 0; et < 8; ++et) {
      const bf16* vp = &VT[et * 16 + m][hh * 8];
      v8bf lo = *(const v8bf*)vp;
      v8bf hi = *(const v8bf*)(vp + 16);
      v16bf vfr;
#pragma unroll
      for (int t = 0; t < 8; ++t) { vfr[t] = lo[t]; vfr[8 + t] = hi[t]; }
      acc[et] = wmma_bf16(pa, vfr, acc[et]);
    }
  }

  // y[b, i, h*128 + e] as bf16 for the output projection GEMM.
  bf16* ybase = yb + (size_t)b * N * D + (size_t)h * DH;
#pragma unroll
  for (int et = 0; et < 8; ++et)
#pragma unroll
    for (int r = 0; r < 8; ++r)
      ybase[(size_t)(i0 + hh * 8 + r) * D + et * 16 + m] = (bf16)acc[et][r];
}

// ---------------------------------------------------------------------------
// K5: out[b,n,o] = sum_d y[b,n,d] * Wout[o,d]  (M=8192, N=1024, K=1024),
// register-tiled 32x64 per wave like K2.
// ---------------------------------------------------------------------------
__global__ void __launch_bounds__(256)
outproj_kernel(const bf16* __restrict__ yb, const bf16* __restrict__ wob,
               float* __restrict__ out) {
  const int lane = threadIdx.x & 31;
  int gw = blockIdx.x * 8 + (threadIdx.x >> 5);
  const int cg = gw & 15; gw >>= 4;   // 64-wide col group (OUT/64 = 16)
  const int rt = gw;                  // 32-tall row tile  (8192/32 = 256)
  const int row0 = rt * 32, n0 = cg * 64;

  v8f c[2][4];
#pragma unroll
  for (int mi = 0; mi < 2; ++mi)
#pragma unroll
    for (int ni = 0; ni < 4; ++ni) c[mi][ni] = zero8();

  for (int kc = 0; kc < D / 32; ++kc) {
    if (kc + 1 < D / 32)
      __builtin_prefetch(yb + (size_t)row0 * D + (kc + 1) * 32, 0, 0);
    v16bf a0 = load_a_frag(yb, D, row0,      kc * 32, lane);
    v16bf a1 = load_a_frag(yb, D, row0 + 16, kc * 32, lane);
#pragma unroll
    for (int ni = 0; ni < 4; ++ni) {
      v16bf bb = load_b_frag(wob, D, n0 + ni * 16, kc * 32, lane);
      c[0][ni] = wmma_bf16(a0, bb, c[0][ni]);
      c[1][ni] = wmma_bf16(a1, bb, c[1][ni]);
    }
  }
  const int m = lane & 15, hh = lane >> 4;
#pragma unroll
  for (int mi = 0; mi < 2; ++mi)
#pragma unroll
    for (int ni = 0; ni < 4; ++ni)
#pragma unroll
      for (int r = 0; r < 8; ++r)
        out[(size_t)(row0 + mi * 16 + hh * 8 + r) * OUT + n0 + ni * 16 + m] =
            c[mi][ni][r];
}

// ---------------------------------------------------------------------------
// Host launcher
// ---------------------------------------------------------------------------
extern "C" void kernel_launch(void* const* d_in, const int* in_sizes, int n_in,
                              void* d_out, int out_size, void* d_ws, size_t ws_size,
                              hipStream_t stream) {
  (void)in_sizes; (void)n_in; (void)out_size; (void)ws_size;
  const float* x    = (const float*)d_in[0];
  const int*   mask = (const int*)  d_in[1];
  const float* Wq   = (const float*)d_in[2];
  const float* Wk   = (const float*)d_in[3];
  const float* Wv   = (const float*)d_in[4];
  const float* Wo   = (const float*)d_in[5];

  float* out  = (float*)d_out;
  float* aout = out + (size_t)BS * N * OUT;   // A[:, H-1] region

  char* ws = (char*)d_ws;
  size_t off = 0;
  auto take = [&](size_t bytes) { char* p = ws + off; off += (bytes + 255) & ~(size_t)255; return p; };
  bf16*  xb   = (bf16*)take((size_t)BS * N * D * 2);
  bf16*  wqb  = (bf16*)take((size_t)H * DH * D * 2);
  bf16*  wkb  = (bf16*)take((size_t)H * DH * D * 2);
  bf16*  wvb  = (bf16*)take((size_t)H * DH * D * 2);
  bf16*  wob  = (bf16*)take((size_t)OUT * D * 2);
  float* ap   = (float*)take((size_t)N * N * 4);
  bf16*  Qb   = (bf16*)take((size_t)BS * H * N * DH * 2);
  bf16*  Kb   = (bf16*)take((size_t)BS * H * N * DH * 2);
  bf16*  Vb   = (bf16*)take((size_t)BS * H * N * DH * 2);
  float* Mrow = (float*)take((size_t)BS * H * N * 4);
  float* Lrow = (float*)take((size_t)BS * H * N * 4);
  bf16*  yb   = (bf16*)take((size_t)BS * N * D * 2);

  convert_kernel<<<1024, 256, 0, stream>>>(x, Wq, Wk, Wv, Wo, xb, wqb, wkb, wvb, wob);
  ap_kernel<<<4096, 256, 0, stream>>>(ap);
  // 3 * 4 * 8 * 64 * 2 = 12288 wave-tiles (32x64 each) / 8 waves per block
  qkv_proj_kernel<<<1536, 256, 0, stream>>>(xb, wqb, wkb, wvb, Qb, Kb, Vb);
  pass1_stats_kernel<<<512, 256, 0, stream>>>(Qb, Kb, ap, mask, Mrow, Lrow);
  pass2_attn_kernel<<<512, 256, 0, stream>>>(Qb, Kb, Vb, ap, mask, Mrow, Lrow, yb, aout);
  // 256 * 16 = 4096 wave-tiles (32x64 each) / 8 waves per block
  outproj_kernel<<<512, 256, 0, stream>>>(yb, wob, out);
}
